// AttentionOutputModule_58299886076271
// MI455X (gfx1250) — matrix operand order repaired and checked by reading, not compiled, not run on hardware
//
#include <hip/hip_runtime.h>
#include <math.h>

// ---------------- problem constants (match reference) ----------------
#define NN    20000
#define EE    600000
#define DD    512
#define HH    8
#define HIDN  1024
#define FINW  536      // real forces-MLP input width
#define FINP  544      // padded to multiple of 32 for WMMA K loop
#define SCALE_ 0.125f  // 1/sqrt(64)

typedef __attribute__((ext_vector_type(16))) __bf16 v16bf;
typedef __attribute__((ext_vector_type(8)))  float  v8f;
typedef __attribute__((ext_vector_type(4)))  unsigned int u32x4;
typedef __attribute__((ext_vector_type(4)))  int    i32x4;
typedef __attribute__((ext_vector_type(8)))  int    i32x8;
typedef __attribute__((ext_vector_type(4)))  float  f32x4;

// ---------------- small device helpers ----------------
__device__ __forceinline__ float bf2f(unsigned short b) {
    return __uint_as_float(((unsigned int)b) << 16);
}
__device__ __forceinline__ unsigned short f2bf(float f) {
    unsigned int u = __float_as_uint(f);
    u += 0x7FFFu + ((u >> 16) & 1u);           // round-to-nearest-even
    return (unsigned short)(u >> 16);
}
// order-preserving float<->uint encoding for deterministic atomic max
__device__ __forceinline__ unsigned int fenc(float x) {
    unsigned int u = __float_as_uint(x);
    return (u & 0x80000000u) ? ~u : (u | 0x80000000u);
}
__device__ __forceinline__ float fdec(unsigned int e) {
    unsigned int u = (e & 0x80000000u) ? (e & 0x7FFFFFFFu) : ~e;
    return __uint_as_float(u);
}
__device__ __forceinline__ float gelu_tanh(float x) {   // jax.nn.gelu (approximate)
    float c = 0.7978845608028654f * (x + 0.044715f * x * x * x);
    return 0.5f * x * (1.f + tanhf(c));
}

// ---------------- weight conversion + swizzle to TDM/WMMA-friendly layout ----------------
// Bt[(k/32)*(Nw*32) + n*32 + (k%32)] = bf16(W[k*Nw + n])   (zero beyond validK rows)
// => each 32(K) x 64(N) B tile is one contiguous 4KB block, already in the
//    [n][k] order the 16x16x32 bf16 B-fragments read from LDS.
__global__ void cvt_swizzleB_kernel(const float* __restrict__ W, unsigned short* __restrict__ Bt,
                                    int Kp, int Nw, int validK) {
    int j = blockIdx.x * blockDim.x + threadIdx.x;
    if (j >= Kp * Nw) return;
    int slab = Nw * 32;
    int kb = j / slab, rem = j % slab;
    int n = rem >> 5, kk = rem & 31;
    int k = kb * 32 + kk;
    Bt[j] = (k < validK) ? f2bf(W[(size_t)k * Nw + n]) : (unsigned short)0;
}

// ---------------- dual LayerNorm: one wave32 per row ----------------
__global__ void ln_kernel(const float* __restrict__ x,
                          const float* __restrict__ ga, const float* __restrict__ ba,
                          const float* __restrict__ gm, const float* __restrict__ bm,
                          unsigned short* __restrict__ zatt,   // [N,512] bf16
                          unsigned short* __restrict__ fin,    // [N,544] bf16 (cols 0..511 here)
                          int n) {
    int row  = blockIdx.x * (blockDim.x >> 5) + (threadIdx.x >> 5);
    int lane = threadIdx.x & 31;
    if (row >= n) return;
    const f32x4* xp = (const f32x4*)(x + (size_t)row * DD + lane * 16);
    float vv[16];
    float s = 0.f, sq = 0.f;
    #pragma unroll
    for (int i = 0; i < 4; ++i) {
        f32x4 t = xp[i];
        #pragma unroll
        for (int j = 0; j < 4; ++j) {
            float v = t[j];
            vv[i * 4 + j] = v;
            s += v; sq += v * v;
        }
    }
    #pragma unroll
    for (int m = 16; m; m >>= 1) { s += __shfl_xor(s, m, 32); sq += __shfl_xor(sq, m, 32); }
    float mean = s * (1.f / DD);
    float var  = sq * (1.f / DD) - mean * mean;
    float rstd = rsqrtf(var + 1e-5f);
    #pragma unroll
    for (int i = 0; i < 16; ++i) {
        int c = lane * 16 + i;
        float xn = (vv[i] - mean) * rstd;
        zatt[(size_t)row * DD   + c] = f2bf(xn * ga[c] + ba[c]);
        fin [(size_t)row * FINP + c] = f2bf(xn * gm[c] + bm[c]);
    }
    if (lane < FINP - FINW)   // zero pad cols 536..543
        fin[(size_t)row * FINP + FINW + lane] = 0;
}

// ---------------- init per-call scratch for edge phase ----------------
__global__ void init_edge_buffers(unsigned int* __restrict__ mxe, float* __restrict__ den,
                                  float* __restrict__ attacc, int n8, int n24) {
    int i = blockIdx.x * blockDim.x + threadIdx.x;
    if (i < n8)  { mxe[i] = fenc(-__builtin_inff()); den[i] = 0.f; }
    if (i < n24) attacc[i] = 0.f;
}

// ---------------- WMMA bf16 GEMM: C = act(A@B + bias) (+res) ----------------
// Block: 256 threads (8 waves), tile 128(M) x 64(N), K-step 32.
// A staged global->regs->LDS (software pipelined); B tile (pre-swizzled, contiguous
// 4KB) moved to LDS by the Tensor Data Mover, fenced with s_wait_tensorcnt.
__global__ __launch_bounds__(256)
void gemm_wmma_bf16(const unsigned short* __restrict__ A, int lda,
                    const unsigned short* __restrict__ Bt, int Nw,  // swizzled B, width Nw
                    const float* __restrict__ bias,
                    const unsigned short* __restrict__ res,   // residual (bf16, stride ldo) or null
                    unsigned short* __restrict__ outBf,       // bf16 out or null
                    float* __restrict__ outF,                 // f32 out or null
                    int ldo, int M, int K, int act) {
    extern __shared__ char smem[];                       // dynamic => LDS offset 0
    unsigned short* As = (unsigned short*)smem;          // [128][32]  (8 KB)
    unsigned short* Bs = (unsigned short*)(smem + 8192); // [64][32]   (4 KB), [n][k]
    u32x4* As4 = (u32x4*)As;
    u32x4* Bs4 = (u32x4*)Bs;

    const int tid  = threadIdx.x;
    const int lane = tid & 31;
    const int wave = tid >> 5;
    const int l    = lane & 15;
    const int hi   = lane >> 4;
    const int mBase = blockIdx.y * 128;
    const int nBase = blockIdx.x * 64;

    const int ar = tid >> 1, ac = (tid & 1) * 16;        // A staging: 32B per thread
    const int aRow = mBase + ar;

    v8f acc[4] = {};

    // pipeline prologue: A slab k0=0 into registers
    u32x4 a0 = (u32x4)0u, a1 = (u32x4)0u;
    if (aRow < M) {
        const u32x4* gp = (const u32x4*)(A + (size_t)aRow * lda + ac);
        a0 = gp[0]; a1 = gp[1];
    }
    unsigned long long gB = (unsigned long long)(const char*)(Bt + (size_t)nBase * 32);
    const unsigned long long slabBytes = (unsigned long long)Nw * 64ull;

    for (int k0 = 0; k0 < K; k0 += 32) {
        __syncthreads();     // previous iteration's fragment reads done
        if (wave == 0) {
            // --- Tensor Data Mover: async 4KB contiguous tile -> LDS @8192 ---
            u32x4 g0; i32x8 g1; i32x4 g2 = (i32x4)0, g3 = (i32x4)0;
            g0[0] = 1u;                                   // count=1, user descriptor
            g0[1] = 8192u;                                // lds_addr (bytes)
            g0[2] = (unsigned int)gB;                     // global_addr[31:0]
            g0[3] = (unsigned int)(gB >> 32) | 0x80000000u; // addr[56:32] | type=2<<30
            g1[0] = 1 << 16;                              // data_size = 2 bytes
            g1[1] = (2048 & 0xFFFF) << 16;                // tensor_dim0 = 2048 (lo16)
            g1[2] = 0x10000;                              // dim0 hi = 0, tensor_dim1 = 1
            g1[3] = 2048 << 16;                           // tile_dim0 = 2048 elements
            g1[4] = 1;                                    // tile_dim1 = 1
            g1[5] = 2048;                                 // tensor_dim0_stride
            g1[6] = 0; g1[7] = 0;
#if __clang_major__ >= 23
            i32x8 g4 = (i32x8)0;
            __builtin_amdgcn_tensor_load_to_lds(g0, g1, g2, g3, g4, 0);
#else
            __builtin_amdgcn_tensor_load_to_lds(g0, g1, g2, g3, 0);
#endif
        }
        // stage A registers -> LDS
        {
            int o = ar * 4 + (ac >> 3);
            As4[o] = a0; As4[o + 1] = a1;
        }
        // pipeline: load next A slab into registers (overlaps with WMMA below)
        if (k0 + 32 < K) {
            a0 = (u32x4)0u; a1 = (u32x4)0u;
            if (aRow < M) {
                const u32x4* gp = (const u32x4*)(A + (size_t)aRow * lda + (k0 + 32) + ac);
                a0 = gp[0]; a1 = gp[1];
            }
            if (k0 + 64 < K)   // gfx1250 global_prefetch_b8 two slabs ahead
                __builtin_prefetch(A + (size_t)aRow * lda + (k0 + 64) + ac, 0, 1);
        }
        if (wave == 0) __builtin_amdgcn_s_wait_tensorcnt(0);  // TDM tile landed
        __syncthreads();

        // A fragment: lane l -> M = wave*16 + l ; elems 0-15: K=hi*16..hi*16+15
        union { v16bf v; u32x4 u[2]; } af;
        {
            int r = wave * 16 + l;
            af.u[0] = As4[r * 4 + hi * 2];
            af.u[1] = As4[r * 4 + hi * 2 + 1];
        }
        // preload all 4 B fragments, then issue WMMAs back-to-back
        union { v16bf v; u32x4 u[2]; } bf0, bf1, bf2, bf3;
        bf0.u[0] = Bs4[(0 * 16 + l) * 4 + hi * 2]; bf0.u[1] = Bs4[(0 * 16 + l) * 4 + hi * 2 + 1];
        bf1.u[0] = Bs4[(1 * 16 + l) * 4 + hi * 2]; bf1.u[1] = Bs4[(1 * 16 + l) * 4 + hi * 2 + 1];
        bf2.u[0] = Bs4[(2 * 16 + l) * 4 + hi * 2]; bf2.u[1] = Bs4[(2 * 16 + l) * 4 + hi * 2 + 1];
        bf3.u[0] = Bs4[(3 * 16 + l) * 4 + hi * 2]; bf3.u[1] = Bs4[(3 * 16 + l) * 4 + hi * 2 + 1];
        acc[0] = __builtin_amdgcn_wmma_f32_16x16x32_bf16(false, af.v, false, bf0.v, (short)0, acc[0], false, false);
        acc[1] = __builtin_amdgcn_wmma_f32_16x16x32_bf16(false, af.v, false, bf1.v, (short)0, acc[1], false, false);
        acc[2] = __builtin_amdgcn_wmma_f32_16x16x32_bf16(false, af.v, false, bf2.v, (short)0, acc[2], false, false);
        acc[3] = __builtin_amdgcn_wmma_f32_16x16x32_bf16(false, af.v, false, bf3.v, (short)0, acc[3], false, false);

        gB += slabBytes;
    }

    // epilogue: lane (l,hi), vgpr r -> C[M = wave*16 + hi*8 + r][N = j*16 + l]
    const int rowBase = mBase + wave * 16 + hi * 8;
    #pragma unroll
    for (int j = 0; j < 4; ++j) {
        const int colG = nBase + j * 16 + l;
        const float bs = bias ? bias[colG] : 0.f;
        #pragma unroll
        for (int r = 0; r < 8; ++r) {
            int rowG = rowBase + r;
            if (rowG < M) {
                float v = acc[j][r] + bs;
                if (act) v = gelu_tanh(v);
                if (res)   v += bf2f(res[(size_t)rowG * ldo + colG]);
                if (outBf) outBf[(size_t)rowG * ldo + colG] = f2bf(v);
                if (outF)  outF [(size_t)rowG * ldo + colG] = v;
            }
        }
    }
}

// ---------------- out-projection (N = 1 or 3): one wave per (row, outcol) ----------------
__global__ void outproj_kernel(const unsigned short* __restrict__ h, int ldh,
                               const float* __restrict__ W, const float* __restrict__ b,
                               float* __restrict__ out, int M, int OD, int K) {
    int gw   = (blockIdx.x * blockDim.x + threadIdx.x) >> 5;
    int lane = threadIdx.x & 31;
    if (gw >= M * OD) return;
    int row = gw / OD, od = gw % OD;
    const unsigned short* hp = h + (size_t)row * ldh;
    float s = 0.f;
    for (int i = lane; i < K; i += 32)
        s += bf2f(hp[i]) * W[(size_t)i * OD + od];
    #pragma unroll
    for (int m = 16; m; m >>= 1) s += __shfl_xor(s, m, 32);
    if (lane == 0) out[(size_t)row * OD + od] = s + b[od];
}

// ---------------- edge attention: logits + segment max ----------------
__global__ void edge_logits_kernel(const float* __restrict__ qf, const float* __restrict__ kf,
                                   const int* __restrict__ row, const int* __restrict__ col,
                                   const float* __restrict__ att_bias,
                                   float* __restrict__ logit, unsigned int* __restrict__ mxe, int E) {
    int e    = (blockIdx.x * blockDim.x + threadIdx.x) >> 5;
    int lane = threadIdx.x & 31;
    if (e >= E) return;
    int r = row[e], c = col[e];
    const f32x4* qp = (const f32x4*)(qf + (size_t)r * DD + lane * 16);
    const f32x4* kp = (const f32x4*)(kf + (size_t)c * DD + lane * 16);
    float s = 0.f;
    #pragma unroll
    for (int i = 0; i < 4; ++i) {
        f32x4 a = qp[i], b = kp[i];
        s += a[0] * b[0] + a[1] * b[1] + a[2] * b[2] + a[3] * b[3];
    }
    // 4 lanes per head (HD=64 = 4 lanes * 16 elems)
    s += __shfl_xor(s, 1, 32);
    s += __shfl_xor(s, 2, 32);
    if ((lane & 3) == 0) {
        int hh = lane >> 2;
        float lg = s * SCALE_ + att_bias[e];
        logit[(size_t)e * HH + hh] = lg;
        atomicMax(&mxe[(size_t)r * HH + hh], fenc(lg));
    }
}

__global__ void node_prep_kernel(const unsigned int* __restrict__ mxe, float* __restrict__ mxf, int n8) {
    int i = blockIdx.x * blockDim.x + threadIdx.x;
    if (i < n8) {
        float m = fdec(mxe[i]);
        if (!(m > -3.0e38f && m < 3.0e38f)) m = 0.f;   // jnp.where(isfinite(mx), mx, 0)
        mxf[i] = m;
    }
}

__global__ void edge_exp_kernel(const int* __restrict__ row, float* __restrict__ logit,
                                const float* __restrict__ mxf, float* __restrict__ den, int e8) {
    int i = blockIdx.x * blockDim.x + threadIdx.x;
    if (i >= e8) return;
    int e = i >> 3, hh = i & 7;
    int r = row[e];
    float p = __expf(logit[i] - mxf[(size_t)r * HH + hh]);
    logit[i] = p;
    atomicAdd(&den[(size_t)r * HH + hh], p);
}

__global__ void edge_contrib_kernel(const int* __restrict__ row, const int* __restrict__ col,
                                    const float* __restrict__ logit, const float* __restrict__ den,
                                    const float* __restrict__ pos, float* __restrict__ attacc, int e8) {
    int i = blockIdx.x * blockDim.x + threadIdx.x;
    if (i >= e8) return;
    int e = i >> 3, hh = i & 7;
    int r = row[e], c = col[e];
    float w = logit[i] / den[(size_t)r * HH + hh];
    float* dst = &attacc[(size_t)r * 24 + hh * 3];
    atomicAdd(dst + 0, w * pos[(size_t)c * 3 + 0]);
    atomicAdd(dst + 1, w * pos[(size_t)c * 3 + 1]);
    atomicAdd(dst + 2, w * pos[(size_t)c * 3 + 2]);
}

__global__ void finalize_att_kernel(const float* __restrict__ attacc, const float* __restrict__ pos,
                                    unsigned short* __restrict__ fin, int n24) {
    int i = blockIdx.x * blockDim.x + threadIdx.x;
    if (i >= n24) return;
    int n = i / 24, j = i % 24, c = j % 3;
    fin[(size_t)n * FINP + DD + j] = f2bf(attacc[i] - pos[(size_t)n * 3 + c]);
}

// ---------------- host launch ----------------
extern "C" void kernel_launch(void* const* d_in, const int* in_sizes, int n_in,
                              void* d_out, int out_size, void* d_ws, size_t ws_size,
                              hipStream_t stream) {
    const float* x        = (const float*)d_in[0];
    const int*   eidx     = (const int*)  d_in[1];
    const float* att_bias = (const float*)d_in[2];
    const float* pos      = (const float*)d_in[3];
    const float* g_att    = (const float*)d_in[5];
    const float* b_att    = (const float*)d_in[6];
    const float* g_mlp    = (const float*)d_in[7];
    const float* b_mlp    = (const float*)d_in[8];
    const float* Wq  = (const float*)d_in[9],  *bq = (const float*)d_in[10];
    const float* Wk  = (const float*)d_in[11], *bk = (const float*)d_in[12];
    const float* Wei = (const float*)d_in[13], *bei = (const float*)d_in[14];
    const float* Weh = (const float*)d_in[15], *beh = (const float*)d_in[16];
    const float* Weo = (const float*)d_in[17], *beo = (const float*)d_in[18];
    const float* Wfi = (const float*)d_in[19], *bfi = (const float*)d_in[20];
    const float* Wfh = (const float*)d_in[21], *bfh = (const float*)d_in[22];
    const float* Wfo = (const float*)d_in[23], *bfo = (const float*)d_in[24];
    float* out = (float*)d_out;

    const int* erow = eidx;
    const int* ecol = eidx + EE;

    // ---- carve workspace ----
    char* w = (char*)d_ws;
    size_t off = 0;
    auto alloc = [&](size_t bytes) -> char* {
        char* p = w + off;
        off = (off + bytes + 255) & ~(size_t)255;
        return p;
    };
    unsigned short* zatt  = (unsigned short*)alloc((size_t)NN * DD   * 2);
    unsigned short* fin   = (unsigned short*)alloc((size_t)NN * FINP * 2);
    float*          qf    = (float*)alloc((size_t)NN * DD * 4);
    float*          kf    = (float*)alloc((size_t)NN * DD * 4);
    float*          logit = (float*)alloc((size_t)EE * HH * 4);
    unsigned int*   mxe   = (unsigned int*)alloc((size_t)NN * HH * 4);
    float*          mxf   = (float*)alloc((size_t)NN * HH * 4);
    float*          den   = (float*)alloc((size_t)NN * HH * 4);
    float*          attac = (float*)alloc((size_t)NN * 24 * 4);
    unsigned short* hA    = (unsigned short*)alloc((size_t)NN * HIDN * 2);
    unsigned short* hB    = (unsigned short*)alloc((size_t)NN * HIDN * 2);
    unsigned short* wq_b  = (unsigned short*)alloc((size_t)DD * DD * 2);
    unsigned short* wk_b  = (unsigned short*)alloc((size_t)DD * DD * 2);
    unsigned short* wei_b = (unsigned short*)alloc((size_t)DD * HIDN * 2);
    unsigned short* weh_b = (unsigned short*)alloc((size_t)HIDN * HIDN * 2);
    unsigned short* wfi_b = (unsigned short*)alloc((size_t)FINP * HIDN * 2);
    unsigned short* wfh_b = (unsigned short*)alloc((size_t)HIDN * HIDN * 2);

    auto cdiv = [](int a, int b) { return (a + b - 1) / b; };

    // ---- weight convert + swizzle into contiguous-TDM-tile layout ----
    cvt_swizzleB_kernel<<<cdiv(DD * DD, 256), 256, 0, stream>>>(Wq, wq_b, DD, DD, DD);
    cvt_swizzleB_kernel<<<cdiv(DD * DD, 256), 256, 0, stream>>>(Wk, wk_b, DD, DD, DD);
    cvt_swizzleB_kernel<<<cdiv(DD * HIDN, 256), 256, 0, stream>>>(Wei, wei_b, DD, HIDN, DD);
    cvt_swizzleB_kernel<<<cdiv(HIDN * HIDN, 256), 256, 0, stream>>>(Weh, weh_b, HIDN, HIDN, HIDN);
    cvt_swizzleB_kernel<<<cdiv(HIDN * HIDN, 256), 256, 0, stream>>>(Wfh, wfh_b, HIDN, HIDN, HIDN);
    cvt_swizzleB_kernel<<<cdiv(FINP * HIDN, 256), 256, 0, stream>>>(Wfi, wfi_b, FINP, HIDN, FINW);

    // ---- layernorms + edge scratch init ----
    ln_kernel<<<cdiv(NN, 8), 256, 0, stream>>>(x, g_att, b_att, g_mlp, b_mlp, zatt, fin, NN);
    init_edge_buffers<<<cdiv(NN * 24, 256), 256, 0, stream>>>(mxe, den, attac, NN * HH, NN * 24);

    const size_t SMEM = 12288;   // 8KB A + 4KB B

    // ---- q/k projections (WMMA) ----
    dim3 gqk(DD / 64, cdiv(NN, 128));
    gemm_wmma_bf16<<<gqk, 256, SMEM, stream>>>(zatt, DD, wq_b, DD, bq, nullptr, nullptr, qf, DD, NN, DD, 0);
    gemm_wmma_bf16<<<gqk, 256, SMEM, stream>>>(zatt, DD, wk_b, DD, bk, nullptr, nullptr, kf, DD, NN, DD, 0);

    // ---- edge softmax-weighted position aggregation ----
    edge_logits_kernel<<<cdiv(EE * 32, 256), 256, 0, stream>>>(qf, kf, erow, ecol, att_bias, logit, mxe, EE);
    node_prep_kernel<<<cdiv(NN * HH, 256), 256, 0, stream>>>(mxe, mxf, NN * HH);
    edge_exp_kernel<<<cdiv(EE * HH, 256), 256, 0, stream>>>(erow, logit, mxf, den, EE * HH);
    edge_contrib_kernel<<<cdiv(EE * HH, 256), 256, 0, stream>>>(erow, ecol, logit, den, pos, attac, EE * HH);
    finalize_att_kernel<<<cdiv(NN * 24, 256), 256, 0, stream>>>(attac, pos, fin, NN * 24);

    // ---- energy ResMLP (WMMA) ----
    dim3 ghid(HIDN / 64, cdiv(NN, 128));
    gemm_wmma_bf16<<<ghid, 256, SMEM, stream>>>(fin, FINP, wei_b, HIDN, bei, nullptr, hA, nullptr, HIDN, NN, DD, 1);
    gemm_wmma_bf16<<<ghid, 256, SMEM, stream>>>(hA, HIDN, weh_b, HIDN, beh, hA, hB, nullptr, HIDN, NN, HIDN, 1);
    outproj_kernel<<<cdiv(NN * 1 * 32, 256), 256, 0, stream>>>(hB, HIDN, Weo, beo, out, NN, 1, HIDN);

    // ---- forces ResMLP (WMMA), reusing hA/hB ----
    gemm_wmma_bf16<<<ghid, 256, SMEM, stream>>>(fin, FINP, wfi_b, HIDN, bfi, nullptr, hA, nullptr, HIDN, NN, FINP, 1);
    gemm_wmma_bf16<<<ghid, 256, SMEM, stream>>>(hA, HIDN, wfh_b, HIDN, bfh, hA, hB, nullptr, HIDN, NN, HIDN, 1);
    outproj_kernel<<<cdiv(NN * 3 * 32, 256), 256, 0, stream>>>(hB, HIDN, Wfo, bfo, out + NN, NN, 3, HIDN);

    (void)in_sizes; (void)n_in; (void)out_size; (void)ws_size;
}
